// TypedCapacityDomainMoEFFN_82145544504121
// MI455X (gfx1250) — matrix operand order repaired and checked by reading, not compiled, not run on hardware
//
#include <hip/hip_runtime.h>
#include <math.h>

// Problem constants (from reference)
#define B_   32
#define T_   1024
#define NTOK (B_ * T_)   // 32768
#define D_   512
#define FF_  2048
#define E_   8
#define CAP_ 5120        // ceil(1.25 * N / E)

// GEMM tiling: 128x256 output tile, K-step 32, 256 threads = 8 waves (2M x 4N),
// each wave owns a 64x64 accumulator block = 4x4 WMMA tiles.
#define BM 128
#define BN 256
#define BK 32
#define NTHREADS 256

typedef __attribute__((ext_vector_type(4)))  float  v4f;
typedef __attribute__((ext_vector_type(8)))  float  v8f;
typedef __attribute__((ext_vector_type(4)))  __bf16 v4bf;
typedef __attribute__((ext_vector_type(8)))  __bf16 v8bf;
typedef __attribute__((ext_vector_type(16))) __bf16 v16bf;

__device__ __forceinline__ float gelu_exact(float x) {
    return 0.5f * x * (1.0f + erff(x * 0.70710678118654752f));
}

// ---------------------------------------------------------------------------
// Router: logits = x @ Wr + br ; gate = softmax prob of argmax ; expert = argmax
// ---------------------------------------------------------------------------
__global__ void router_kernel(const float* __restrict__ X,
                              const float* __restrict__ Wr,   // [D, E]
                              const float* __restrict__ br,   // [E]
                              float* __restrict__ gate,       // [N]
                              int*   __restrict__ expsel)     // [N]
{
    int i = blockIdx.x * blockDim.x + threadIdx.x;
    if (i >= NTOK) return;
    float acc[E_];
#pragma unroll
    for (int e = 0; e < E_; ++e) acc[e] = br[e];
    const float* xr = X + (size_t)i * D_;
    for (int k = 0; k < D_; ++k) {
        float xv = xr[k];
        const float* wr = Wr + k * E_;
#pragma unroll
        for (int e = 0; e < E_; ++e) acc[e] += xv * wr[e];
    }
    float m = acc[0]; int am = 0;
#pragma unroll
    for (int e = 1; e < E_; ++e) { if (acc[e] > m) { m = acc[e]; am = e; } }
    float s = 0.0f;
#pragma unroll
    for (int e = 0; e < E_; ++e) s += __expf(acc[e] - m);
    gate[i]   = 1.0f / s;    // prob of the argmax
    expsel[i] = am;
}

// ---------------------------------------------------------------------------
// Deterministic capacity scan (one 1024-thread workgroup)
// ---------------------------------------------------------------------------
#define SCAN_THREADS 1024
#define TOKS_PER_THREAD (NTOK / SCAN_THREADS)  // 32

__global__ void scan_kernel(const int* __restrict__ expsel,   // [N]
                            int* __restrict__ slotmap,        // [E*CAP]: slot -> token
                            int* __restrict__ counts)         // [E] (clamped to CAP)
{
    __shared__ int hist[SCAN_THREADS * E_];    // 32 KB
    int t = threadIdx.x;
    int cnt[E_];
#pragma unroll
    for (int e = 0; e < E_; ++e) cnt[e] = 0;
    int base = t * TOKS_PER_THREAD;
    for (int j = 0; j < TOKS_PER_THREAD; ++j) cnt[expsel[base + j]]++;
#pragma unroll
    for (int e = 0; e < E_; ++e) hist[t * E_ + e] = cnt[e];
    __syncthreads();
    if (t < E_) {
        int run = 0;
        for (int i = 0; i < SCAN_THREADS; ++i) {
            int v = hist[i * E_ + t];
            hist[i * E_ + t] = run;
            run += v;
        }
        counts[t] = run < CAP_ ? run : CAP_;
    }
    __syncthreads();
    int off[E_];
#pragma unroll
    for (int e = 0; e < E_; ++e) off[e] = hist[t * E_ + e];
    for (int j = 0; j < TOKS_PER_THREAD; ++j) {
        int tok = base + j;
        int e   = expsel[tok];
        int p   = off[e]++;
        if (p < CAP_) slotmap[e * CAP_ + p] = tok;
    }
}

// ---------------------------------------------------------------------------
// GEMM 1: H = gelu( gather(X) @ W1 + b1 ), bf16 out, f32 accumulate via WMMA
// ---------------------------------------------------------------------------
__global__ __launch_bounds__(NTHREADS)
void ffn1_kernel(const float* __restrict__ X,       // [NTOK, D]
                 const float* __restrict__ W1,      // [D, FF]
                 const float* __restrict__ b1,      // [FF]
                 __bf16* __restrict__ H,            // [>=tile rows, FF]
                 const int* __restrict__ slotmap,   // per-expert base or null
                 const int* __restrict__ counts,    // [E] or null
                 int expertIdx, int rowOffset, int sharedRows)
{
    __shared__ __bf16 Abuf[BM][BK];   //  8 KB
    __shared__ __bf16 Bbuf[BN][BK];   // 16 KB, stored [n][k]

    const int rowbase = blockIdx.y * BM;
    const int nbase   = blockIdx.x * BN;
    const int count   = (slotmap != nullptr) ? counts[expertIdx] : sharedRows;
    if (rowbase >= count) return;

    const int tid  = threadIdx.x;
    // A staging: 8 threads/row, v4f each, 4 row-passes
    const int colA = (tid & 7) * 4;
    const int rA0  = tid >> 3;          // 0..31
    // B staging: thread covers 4 K-rows x 4 N-cols, 2 N-passes
    const int k4   = (tid >> 5) * 4;    // 0,4,..,28
    const int nB0  = (tid & 31) * 4;    // 0..124

    int srcRow[4];
#pragma unroll
    for (int p = 0; p < 4; ++p) {
        int g = rowbase + p * 32 + rA0;
        srcRow[p] = (slotmap != nullptr) ? ((g < count) ? slotmap[g] : -1)
                                         : (rowOffset + g);
    }

    const int lane = tid & 31;
    const int l16  = lane & 15;
    const int half = lane >> 4;
    const int ka0  = half * 8;     // A frag: K chunks {ka0..+7, ka0+16..+23}
    const int kb0  = half * 16;    // unused; B frag reads the same ka0 chunks as A
    const int waveId = tid >> 5;
    const int waveM  = waveId >> 2;  // 0..1 (64 rows each)
    const int waveN  = waveId & 3;   // 0..3 (64 cols each)

    v8f acc[4][4] = {};

    for (int kt = 0; kt < D_ / BK; ++kt) {
        // ---- stage A: f32 -> packed bf16, one b64 store per thread-pass ----
#pragma unroll
        for (int p = 0; p < 4; ++p) {
            int rA = p * 32 + rA0;
            v4f v = {};
            if (srcRow[p] >= 0)
                v = *(const v4f*)(X + (size_t)srcRow[p] * D_ + kt * BK + colA);
            v4bf pk;
            pk[0] = (__bf16)v[0]; pk[1] = (__bf16)v[1];
            pk[2] = (__bf16)v[2]; pk[3] = (__bf16)v[3];
            *(v4bf*)&Abuf[rA][colA] = pk;
        }
        // ---- stage B: 4 coalesced K-rows -> packed transposed b64 stores ----
#pragma unroll
        for (int p = 0; p < 2; ++p) {
            int n0 = nB0 + p * 128;
            const float* wp = W1 + (size_t)(kt * BK + k4) * FF_ + nbase + n0;
            v4f r0 = *(const v4f*)(wp);
            v4f r1 = *(const v4f*)(wp + FF_);
            v4f r2 = *(const v4f*)(wp + 2 * FF_);
            v4f r3 = *(const v4f*)(wp + 3 * FF_);
#pragma unroll
            for (int j = 0; j < 4; ++j) {
                v4bf pk;
                pk[0] = (__bf16)r0[j]; pk[1] = (__bf16)r1[j];
                pk[2] = (__bf16)r2[j]; pk[3] = (__bf16)r3[j];
                *(v4bf*)&Bbuf[n0 + j][k4] = pk;   // Bbuf[n][k4..k4+3]
            }
        }
        if (kt + 1 < D_ / BK)
            __builtin_prefetch(W1 + (size_t)((kt + 1) * BK + k4) * FF_ + nbase + nB0, 0, 1);
        __syncthreads();

        v16bf afrag[4];
#pragma unroll
        for (int i = 0; i < 4; ++i) {
            int r = waveM * 64 + i * 16 + l16;
            v8bf lo = *(const v8bf*)&Abuf[r][ka0];
            v8bf hi = *(const v8bf*)&Abuf[r][ka0 + 16];
            afrag[i] = __builtin_shufflevector(lo, hi, 0,1,2,3,4,5,6,7,8,9,10,11,12,13,14,15);
        }
#pragma unroll
        for (int j = 0; j < 4; ++j) {
            int c = waveN * 64 + j * 16 + l16;
            v8bf lo = *(const v8bf*)&Bbuf[c][ka0];
            v8bf hi = *(const v8bf*)&Bbuf[c][ka0 + 16];
            v16bf bfrag = __builtin_shufflevector(lo, hi, 0,1,2,3,4,5,6,7,8,9,10,11,12,13,14,15);
#pragma unroll
            for (int i = 0; i < 4; ++i)
                acc[i][j] = __builtin_amdgcn_wmma_f32_16x16x32_bf16(
                    false, afrag[i], false, bfrag, (short)0, acc[i][j], false, false);
        }
        __syncthreads();
    }

    // ---- epilogue: bias + exact gelu, store bf16 H ----
#pragma unroll
    for (int i = 0; i < 4; ++i) {
#pragma unroll
        for (int v = 0; v < 8; ++v) {
            int grow = rowbase + waveM * 64 + i * 16 + half * 8 + v;
#pragma unroll
            for (int j = 0; j < 4; ++j) {
                int gcol = nbase + waveN * 64 + j * 16 + l16;
                float val = acc[i][j][v] + b1[gcol];
                H[(size_t)grow * FF_ + gcol] = (__bf16)gelu_exact(val);
            }
        }
    }
}

// ---------------------------------------------------------------------------
// GEMM 2: contrib = H @ W2 + b2 ; expert: Y[tok] += gate*contrib ; shared: Y = contrib
// A operand (H) is already bf16 -> staged with async global->LDS b128 copies.
// ---------------------------------------------------------------------------
__global__ __launch_bounds__(NTHREADS)
void ffn2_kernel(const __bf16* __restrict__ H,      // [>=tile rows, FF] bf16
                 const float* __restrict__ W2,      // [FF, D]
                 const float* __restrict__ b2,      // [D]
                 float* __restrict__ Y,             // [NTOK, D]
                 const int* __restrict__ slotmap,
                 const int* __restrict__ counts,
                 const float* __restrict__ gate,
                 int expertIdx, int rowOffset, int sharedRows)
{
    __shared__ __bf16 Abuf[BM][BK];   //  8 KB
    __shared__ __bf16 Bbuf[BN][BK];   // 16 KB

    const int rowbase = blockIdx.y * BM;
    const int nbase   = blockIdx.x * BN;
    const int count   = (slotmap != nullptr) ? counts[expertIdx] : sharedRows;
    if (rowbase >= count) return;

    const int tid  = threadIdx.x;
    // A staging via async b128: thread covers 8 bf16 (16B), 2 row-passes
    const int colA8 = (tid & 3) * 8;    // 0,8,16,24
    const int rAa   = tid >> 2;         // 0..63
    // B staging: 4 K-rows x 4 N-cols, 2 N-passes
    const int k4  = (tid >> 5) * 4;
    const int nB0 = (tid & 31) * 4;

    const int lane = tid & 31;
    const int l16  = lane & 15;
    const int half = lane >> 4;
    const int ka0  = half * 8;
    const int kb0  = half * 16;
    const int waveId = tid >> 5;
    const int waveM  = waveId >> 2;
    const int waveN  = waveId & 3;

    // precompute LDS byte addresses for the async copies (low 32 bits of the
    // generic pointer to LDS == wave-relative LDS byte offset)
    unsigned ldsA0 = (unsigned)(uintptr_t)&Abuf[rAa][colA8];
    unsigned ldsA1 = (unsigned)(uintptr_t)&Abuf[64 + rAa][colA8];

    v8f acc[4][4] = {};

    for (int kt = 0; kt < FF_ / BK; ++kt) {
        // ---- stage A: async global -> LDS, 16B per lane per pass ----
        {
            const __bf16* g0 = H + (size_t)(rowbase + rAa) * FF_ + kt * BK + colA8;
            const __bf16* g1 = H + (size_t)(rowbase + 64 + rAa) * FF_ + kt * BK + colA8;
            unsigned long long ga0 = (unsigned long long)(uintptr_t)g0;
            unsigned long long ga1 = (unsigned long long)(uintptr_t)g1;
            asm volatile("global_load_async_to_lds_b128 %0, %1, off"
                         :: "v"(ldsA0), "v"(ga0) : "memory");
            asm volatile("global_load_async_to_lds_b128 %0, %1, off"
                         :: "v"(ldsA1), "v"(ga1) : "memory");
        }
        // ---- stage B: coalesced f32 loads, packed transposed bf16 stores ----
#pragma unroll
        for (int p = 0; p < 2; ++p) {
            int n0 = nB0 + p * 128;
            const float* wp = W2 + (size_t)(kt * BK + k4) * D_ + nbase + n0;
            v4f r0 = *(const v4f*)(wp);
            v4f r1 = *(const v4f*)(wp + D_);
            v4f r2 = *(const v4f*)(wp + 2 * D_);
            v4f r3 = *(const v4f*)(wp + 3 * D_);
#pragma unroll
            for (int j = 0; j < 4; ++j) {
                v4bf pk;
                pk[0] = (__bf16)r0[j]; pk[1] = (__bf16)r1[j];
                pk[2] = (__bf16)r2[j]; pk[3] = (__bf16)r3[j];
                *(v4bf*)&Bbuf[n0 + j][k4] = pk;
            }
        }
        if (kt + 1 < FF_ / BK)
            __builtin_prefetch(W2 + (size_t)((kt + 1) * BK + k4) * D_ + nbase + nB0, 0, 1);
        // wait for this wave's async LDS writes, then sync the workgroup
        asm volatile("s_wait_asynccnt 0x0" ::: "memory");
        __syncthreads();

        v16bf afrag[4];
#pragma unroll
        for (int i = 0; i < 4; ++i) {
            int r = waveM * 64 + i * 16 + l16;
            v8bf lo = *(const v8bf*)&Abuf[r][ka0];
            v8bf hi = *(const v8bf*)&Abuf[r][ka0 + 16];
            afrag[i] = __builtin_shufflevector(lo, hi, 0,1,2,3,4,5,6,7,8,9,10,11,12,13,14,15);
        }
#pragma unroll
        for (int j = 0; j < 4; ++j) {
            int c = waveN * 64 + j * 16 + l16;
            v8bf lo = *(const v8bf*)&Bbuf[c][ka0];
            v8bf hi = *(const v8bf*)&Bbuf[c][ka0 + 16];
            v16bf bfrag = __builtin_shufflevector(lo, hi, 0,1,2,3,4,5,6,7,8,9,10,11,12,13,14,15);
#pragma unroll
            for (int i = 0; i < 4; ++i)
                acc[i][j] = __builtin_amdgcn_wmma_f32_16x16x32_bf16(
                    false, afrag[i], false, bfrag, (short)0, acc[i][j], false, false);
        }
        __syncthreads();
    }

    // ---- epilogue ----
#pragma unroll
    for (int i = 0; i < 4; ++i) {
#pragma unroll
        for (int v = 0; v < 8; ++v) {
            int grow = rowbase + waveM * 64 + i * 16 + half * 8 + v;
            if (slotmap != nullptr) {
                if (grow < count) {
                    int tok = slotmap[grow];
                    float g = gate[tok];
#pragma unroll
                    for (int j = 0; j < 4; ++j) {
                        int gcol = nbase + waveN * 64 + j * 16 + l16;
                        float* o = Y + (size_t)tok * D_ + gcol;
                        *o += g * (acc[i][j][v] + b2[gcol]);   // unique (tok,col) per launch
                    }
                }
            } else {
                int tok = rowOffset + grow;
#pragma unroll
                for (int j = 0; j < 4; ++j) {
                    int gcol = nbase + waveN * 64 + j * 16 + l16;
                    Y[(size_t)tok * D_ + gcol] = acc[i][j][v] + b2[gcol];
                }
            }
        }
    }
}

// ---------------------------------------------------------------------------
extern "C" void kernel_launch(void* const* d_in, const int* in_sizes, int n_in,
                              void* d_out, int out_size, void* d_ws, size_t ws_size,
                              hipStream_t stream) {
    const float* x     = (const float*)d_in[0];
    const float* Wr_sp = (const float*)d_in[1];
    const float* br_sp = (const float*)d_in[2];
    const float* Wr_fq = (const float*)d_in[3];
    const float* br_fq = (const float*)d_in[4];
    const float* Ws1   = (const float*)d_in[5];
    const float* bs1   = (const float*)d_in[6];
    const float* Ws2   = (const float*)d_in[7];
    const float* bs2   = (const float*)d_in[8];
    const float* W1_sp = (const float*)d_in[9];
    const float* b1_sp = (const float*)d_in[10];
    const float* W2_sp = (const float*)d_in[11];
    const float* b2_sp = (const float*)d_in[12];
    const float* W1_fq = (const float*)d_in[13];
    const float* b1_fq = (const float*)d_in[14];
    const float* W2_fq = (const float*)d_in[15];
    const float* b2_fq = (const float*)d_in[16];
    float* out = (float*)d_out;

    // Workspace (~22 MB): routing arrays + one reusable bf16 H buffer.
    char* ws = (char*)d_ws;
    size_t off = 0;
    auto wsalloc = [&](size_t bytes) -> void* {
        void* p = ws + off;
        off += (bytes + 255) & ~(size_t)255;
        return p;
    };
    float*  gate_sp = (float*)wsalloc((size_t)NTOK * 4);
    float*  gate_fq = (float*)wsalloc((size_t)NTOK * 4);
    int*    exp_sp  = (int*)wsalloc((size_t)NTOK * 4);
    int*    exp_fq  = (int*)wsalloc((size_t)NTOK * 4);
    int*    slot_sp = (int*)wsalloc((size_t)E_ * CAP_ * 4);
    int*    slot_fq = (int*)wsalloc((size_t)E_ * CAP_ * 4);
    int*    cnt_sp  = (int*)wsalloc((size_t)E_ * 4);
    int*    cnt_fq  = (int*)wsalloc((size_t)E_ * 4);
    __bf16* H       = (__bf16*)wsalloc((size_t)CAP_ * FF_ * 2);  // 20 MB, reused

    // 1) routing
    router_kernel<<<NTOK / 256, 256, 0, stream>>>(x, Wr_sp, br_sp, gate_sp, exp_sp);
    router_kernel<<<NTOK / 256, 256, 0, stream>>>(x, Wr_fq, br_fq, gate_fq, exp_fq);
    scan_kernel<<<1, SCAN_THREADS, 0, stream>>>(exp_sp, slot_sp, cnt_sp);
    scan_kernel<<<1, SCAN_THREADS, 0, stream>>>(exp_fq, slot_fq, cnt_fq);

    // 2) shared dense FFN in row chunks (write-mode initializes d_out)
    const int CHUNK = 4096;
    for (int c = 0; c < NTOK / CHUNK; ++c) {
        dim3 g1(FF_ / BN, CHUNK / BM);   // (8, 32)
        ffn1_kernel<<<g1, NTHREADS, 0, stream>>>(x, Ws1, bs1, H,
                                                 nullptr, nullptr, 0, c * CHUNK, CHUNK);
        dim3 g2(D_ / BN, CHUNK / BM);    // (2, 32)
        ffn2_kernel<<<g2, NTHREADS, 0, stream>>>(H, Ws2, bs2, out,
                                                 nullptr, nullptr, nullptr, 0, c * CHUNK, CHUNK);
    }

    // 3) expert banks (spatial then spectral), one expert at a time, H reused
    for (int bank = 0; bank < 2; ++bank) {
        const float* W1b = bank ? W1_fq : W1_sp;
        const float* b1b = bank ? b1_fq : b1_sp;
        const float* W2b = bank ? W2_fq : W2_sp;
        const float* b2b = bank ? b2_fq : b2_sp;
        const float* gb  = bank ? gate_fq : gate_sp;
        int*         sl  = bank ? slot_fq : slot_sp;
        int*         cn  = bank ? cnt_fq : cnt_sp;
        for (int e = 0; e < E_; ++e) {
            dim3 g1(FF_ / BN, CAP_ / BM);   // (8, 40)
            ffn1_kernel<<<g1, NTHREADS, 0, stream>>>(
                x, W1b + (size_t)e * D_ * FF_, b1b + (size_t)e * FF_, H,
                sl + (size_t)e * CAP_, cn, e, 0, 0);
            dim3 g2(D_ / BN, CAP_ / BM);    // (2, 40)
            ffn2_kernel<<<g2, NTHREADS, 0, stream>>>(
                H, W2b + (size_t)e * FF_ * D_, b2b + (size_t)e * D_, out,
                sl + (size_t)e * CAP_, cn, gb, e, 0, 0);
        }
    }
}
